// GINENet_3736621547800
// MI455X (gfx1250) — matrix operand order repaired and checked by reading, not compiled, not run on hardware
//
#include <hip/hip_runtime.h>
#include <math.h>

#define D_IN 128
#define BN_EPS_F 1e-5f

typedef __attribute__((ext_vector_type(16))) __bf16 v16bf;
typedef __attribute__((ext_vector_type(8)))  float  v8f;

union BF16x16 { uint4 u[2]; v16bf v; };

// ---------------------------------------------------------------- zero fill
__global__ void k_zero(float* __restrict__ p, long n) {
  long i = (long)blockIdx.x * blockDim.x + threadIdx.x;
  long stride = (long)gridDim.x * blockDim.x;
  for (; i < n; i += stride) p[i] = 0.0f;
}

// --------------------------------------------------- fused GINE message+scatter
// msg = relu(h[src] + ea*lin_w + lin_b)  (e recomputed on the fly, never stored)
// aggr[dst] += msg via f32 HW atomics; h (25.6MB) and aggr (25.6MB) are L2-resident.
// One wave = one edge; lane L handles features 4L..4L+3 (coalesced 512B row).
__global__ void k_edge_scatter(const float* __restrict__ h,
                               const float* __restrict__ ea,
                               const int*   __restrict__ src,
                               const int*   __restrict__ dst,
                               const float* __restrict__ lw,
                               const float* __restrict__ lb,
                               float* __restrict__ aggr, int E) {
  long tid  = (long)blockIdx.x * blockDim.x + threadIdx.x;
  int  edge = (int)(tid >> 5);
  if (edge >= E) return;
  int f = (int)(tid & 31) * 4;
  float a = ea[edge];
  int s = src[edge], d = dst[edge];
  const float4 hv = *(const float4*)(h + (long)s * D_IN + f);
  const float4 wv = *(const float4*)(lw + f);
  const float4 bv = *(const float4*)(lb + f);
  float m0 = fmaxf(fmaf(a, wv.x, hv.x + bv.x), 0.0f);
  float m1 = fmaxf(fmaf(a, wv.y, hv.y + bv.y), 0.0f);
  float m2 = fmaxf(fmaf(a, wv.z, hv.z + bv.z), 0.0f);
  float m3 = fmaxf(fmaf(a, wv.w, hv.w + bv.w), 0.0f);
  float* p = aggr + (long)d * D_IN + f;
  atomicAdd(p + 0, m0);
  atomicAdd(p + 1, m1);
  atomicAdd(p + 2, m2);
  atomicAdd(p + 3, m3);
}

// -------------------------------------------- weight pack: f32 [K,Dout] row-major
// -> bf16 [ncols, K] column-major, zero-padded to ncols = TN*16 columns.
__global__ void k_pack_w(const float* __restrict__ B, __bf16* __restrict__ Bt,
                         int K, int Dout, int ncols) {
  int i = blockIdx.x * blockDim.x + threadIdx.x;
  if (i >= ncols * K) return;
  int col = i / K, k = i % K;
  float v = (col < Dout) ? B[(long)k * Dout + col] : 0.0f;
  Bt[i] = (__bf16)v;
}

// --------------------------------------------- bf16 WMMA GEMM + bias + ReLU + BN stats
// MODE 0: A = A1 + A2         (K = K1)      -- conv residual h + aggr
// MODE 1: A = concat(A1, A2)  (K = K1 + K2) -- [x1 | x2]
// MODE 2: A = A1              (K = K1)
// Fully templated dims: all B-tile offsets become immediate ioffsets off ONE base
// pointer (no per-tile 64-bit induction vars -> no spills). One wave owns one
// 16-row tile and all TN 16-col tiles; TN back-to-back WMMAs per k-step.
template <int MODE, int TN, int K1, int K2, int DOUT>
__global__ void k_gemm_relu_stats(const float* __restrict__ A1,
                                  const float* __restrict__ A2,
                                  const __bf16* __restrict__ Bt,
                                  const float* __restrict__ bias,
                                  float* __restrict__ out,
                                  float* __restrict__ colsum,
                                  float* __restrict__ colsumsq,
                                  int M) {
  constexpr int K = (MODE == 1) ? (K1 + K2) : K1;
  const int lane = threadIdx.x & 31;
  const int wv   = threadIdx.x >> 5;
  const int tile = blockIdx.x * (blockDim.x >> 5) + wv;
  const int tm   = (M + 15) >> 4;
  if (tile >= tm) return;
  const int m0   = tile << 4;
  const int half = lane >> 4;
  const int l16  = lane & 15;
  int row  = m0 + l16;
  int rowc = (row < M) ? row : (M - 1);   // clamp: OOB rows masked at store

  // Single per-lane B base; tile t adds the compile-time offset t*16*K elements.
  const __bf16* bbase = Bt + (long)l16 * K + half * 8;
  const float*  a1row = A1 + (long)rowc * K1;
  const float*  a2row = (MODE == 0) ? (A2 + (long)rowc * K1)
                       : (MODE == 1) ? (A2 + (long)rowc * K2) : nullptr;

  const v8f vzero = {};
  v8f acc[TN];
#pragma unroll
  for (int t = 0; t < TN; ++t) acc[t] = vzero;

#pragma unroll
  for (int k0 = 0; k0 < K; k0 += 32) {
    // ---- A fragment: elems 0..7 = K[k0+8h .. +7], elems 8..15 = K[k0+16+8h .. +7]
    float fa[16];
#pragma unroll
    for (int c = 0; c < 2; ++c) {
      const int ks = k0 + c * 16 + half * 8;   // 8 consecutive K values
      const float* p;
      if (MODE == 1) {
        p = (ks < K1) ? (a1row + ks) : (a2row + (ks - K1));
      } else {
        p = a1row + ks;
      }
      float4 q0 = ((const float4*)p)[0];
      float4 q1 = ((const float4*)p)[1];
      float v0 = q0.x, v1 = q0.y, v2 = q0.z, v3 = q0.w;
      float v4 = q1.x, v5 = q1.y, v6 = q1.z, v7 = q1.w;
      if (MODE == 0) {
        const float* p2 = a2row + ks;
        float4 r0 = ((const float4*)p2)[0];
        float4 r1 = ((const float4*)p2)[1];
        v0 += r0.x; v1 += r0.y; v2 += r0.z; v3 += r0.w;
        v4 += r1.x; v5 += r1.y; v6 += r1.z; v7 += r1.w;
      }
      fa[c * 8 + 0] = v0; fa[c * 8 + 1] = v1; fa[c * 8 + 2] = v2; fa[c * 8 + 3] = v3;
      fa[c * 8 + 4] = v4; fa[c * 8 + 5] = v5; fa[c * 8 + 6] = v6; fa[c * 8 + 7] = v7;
    }
    v16bf af;
#pragma unroll
    for (int e = 0; e < 16; ++e) af[e] = (__bf16)fa[e];

    // ---- TN x (B fragment + WMMA); offsets are compile-time immediates
#pragma unroll
    for (int t = 0; t < TN; ++t) {
      const __bf16* bp = bbase + (t * 16 * K + k0);
      BF16x16 bb;
      bb.u[0] = ((const uint4*)bp)[0];
      bb.u[1] = ((const uint4*)bp)[2];
      acc[t] = __builtin_amdgcn_wmma_f32_16x16x32_bf16(
          false, af, false, bb.v, (short)0, acc[t], false, false);
    }
  }

  // ---- Epilogue: bias + ReLU, store, per-column BN stats (VGPR j -> row m0+8h+j)
#pragma unroll
  for (int t = 0; t < TN; ++t) {
    const int col = t * 16 + l16;
    const bool cok = (col < DOUT);          // folds away when DOUT % 16 == 0
    float bsv = cok ? bias[col] : 0.0f;
    float s = 0.0f, sq = 0.0f;
#pragma unroll
    for (int j = 0; j < 8; ++j) {
      int r = m0 + half * 8 + j;
      float v = fmaxf(acc[t][j] + bsv, 0.0f);
      if (r < M && cok) {
        out[(long)r * DOUT + col] = v;
        s  += v;
        sq += v * v;
      }
    }
    if (cok) {
      atomicAdd(&colsum[col],   s);
      atomicAdd(&colsumsq[col], sq);
    }
  }
}

// -------------------------------------------------------------- BN finalize
__global__ void k_bn_finalize(const float* __restrict__ colsum,
                              const float* __restrict__ colsumsq,
                              const float* __restrict__ g,
                              const float* __restrict__ be,
                              float* __restrict__ scale,
                              float* __restrict__ shift,
                              int Dout, float invN) {
  int c = blockIdx.x * blockDim.x + threadIdx.x;
  if (c >= Dout) return;
  float m   = colsum[c] * invN;
  float var = colsumsq[c] * invN - m * m;      // biased variance
  float sc  = g[c] * rsqrtf(var + BN_EPS_F);
  scale[c]  = sc;
  shift[c]  = be[c] - m * sc;
}

// ---------------------------------------------------------------- BN apply
__global__ void k_bn_apply(const float* __restrict__ pre,
                           const float* __restrict__ scale,
                           const float* __restrict__ shift,
                           float* __restrict__ out, long n, int Dout) {
  long i = (long)blockIdx.x * blockDim.x + threadIdx.x;
  long stride = (long)gridDim.x * blockDim.x;
  for (; i < n; i += stride) {
    int c = (int)(i % Dout);
    out[i] = fmaf(pre[i], scale[c], shift[c]);
  }
}

// ================================================================ host side
static inline int gblocks(long n, int bs) {
  long b = (n + bs - 1) / bs;
  if (b > 1048576L) b = 1048576L;
  return (int)b;
}

extern "C" void kernel_launch(void* const* d_in, const int* in_sizes, int n_in,
                              void* d_out, int out_size, void* d_ws, size_t ws_size,
                              hipStream_t stream) {
  const float* x   = (const float*)d_in[0];
  const float* ea  = (const float*)d_in[1];
  const int*   ei  = (const int*)d_in[2];
  const float* lw  = (const float*)d_in[3];
  const float* lb  = (const float*)d_in[4];
  const float* w1  = (const float*)d_in[5];
  const float* b1  = (const float*)d_in[6];
  const float* g1  = (const float*)d_in[7];
  const float* be1 = (const float*)d_in[8];
  const float* w2  = (const float*)d_in[9];
  const float* b2  = (const float*)d_in[10];
  const float* g2  = (const float*)d_in[11];
  const float* be2 = (const float*)d_in[12];
  const float* w3  = (const float*)d_in[13];
  const float* b3  = (const float*)d_in[14];
  const float* g3  = (const float*)d_in[15];
  const float* be3 = (const float*)d_in[16];
  const float* w4  = (const float*)d_in[17];
  const float* b4  = (const float*)d_in[18];
  const float* g4  = (const float*)d_in[19];
  const float* be4 = (const float*)d_in[20];
  const float* w5  = (const float*)d_in[21];
  const float* b5  = (const float*)d_in[22];
  const float* g5  = (const float*)d_in[23];
  const float* be5 = (const float*)d_in[24];

  const int N = in_sizes[0] / D_IN;
  const int E = in_sizes[1];
  const int C = 10;
  const int* src = ei;
  const int* dst = ei + E;
  const float invN = 1.0f / (float)N;

  // ws layout (floats): aggr[N,128] pre[N,128] x1[N,128] x2[N,64] hA[N,96] hB[N,96]
  //                     stats[512] wpack[16K floats => 32K bf16]
  float* ws   = (float*)d_ws;
  long   NB   = (long)N;
  float* aggr = ws;
  float* pre  = aggr + NB * 128;
  float* x1   = pre  + NB * 128;
  float* x2   = x1   + NB * 128;
  float* hA   = x2   + NB * 64;
  float* hB   = hA   + NB * 96;
  float* st   = hB   + NB * 96;
  float* colsum   = st;
  float* colsumsq = st + 128;
  float* scale    = st + 256;
  float* shift    = st + 384;
  __bf16* wpack   = (__bf16*)(st + 512);

  const int tm      = (N + 15) / 16;
  const int gblk    = (tm + 7) / 8;   // 8 waves (16-row tiles) per 256-thread block
  const long escat  = ((long)E * 32 + 255) / 256;

  // ---------------- conv1: aggr = scatter(relu(x[src] + e)) ----------------
  k_zero<<<gblocks(NB * 128, 256), 256, 0, stream>>>(aggr, NB * 128);
  k_edge_scatter<<<(int)escat, 256, 0, stream>>>(x, ea, src, dst, lw, lb, aggr, E);
  k_zero<<<1, 256, 0, stream>>>(st, 256);
  k_pack_w<<<(128 * 128 + 255) / 256, 256, 0, stream>>>(w1, wpack, 128, 128, 128);
  k_gemm_relu_stats<0, 8, 128, 0, 128><<<gblk, 256, 0, stream>>>(
      x, aggr, wpack, b1, pre, colsum, colsumsq, N);
  k_bn_finalize<<<1, 128, 0, stream>>>(colsum, colsumsq, g1, be1, scale, shift, 128, invN);
  k_bn_apply<<<gblocks(NB * 128, 256), 256, 0, stream>>>(pre, scale, shift, x1, NB * 128, 128);

  // ---------------- conv2: aggr = scatter(relu(x1[src] + e)) ----------------
  k_zero<<<gblocks(NB * 128, 256), 256, 0, stream>>>(aggr, NB * 128);
  k_edge_scatter<<<(int)escat, 256, 0, stream>>>(x1, ea, src, dst, lw, lb, aggr, E);
  k_zero<<<1, 256, 0, stream>>>(st, 256);
  k_pack_w<<<(64 * 128 + 255) / 256, 256, 0, stream>>>(w2, wpack, 128, 64, 64);
  k_gemm_relu_stats<0, 4, 128, 0, 64><<<gblk, 256, 0, stream>>>(
      x1, aggr, wpack, b2, pre, colsum, colsumsq, N);
  k_bn_finalize<<<1, 128, 0, stream>>>(colsum, colsumsq, g2, be2, scale, shift, 64, invN);
  k_bn_apply<<<gblocks(NB * 64, 256), 256, 0, stream>>>(pre, scale, shift, x2, NB * 64, 64);

  // ---------------- lin1: [x1 | x2] @ w3 ----------------
  k_zero<<<1, 256, 0, stream>>>(st, 256);
  k_pack_w<<<(96 * 192 + 255) / 256, 256, 0, stream>>>(w3, wpack, 192, 96, 96);
  k_gemm_relu_stats<1, 6, 128, 64, 96><<<gblk, 256, 0, stream>>>(
      x1, x2, wpack, b3, pre, colsum, colsumsq, N);
  k_bn_finalize<<<1, 128, 0, stream>>>(colsum, colsumsq, g3, be3, scale, shift, 96, invN);
  k_bn_apply<<<gblocks(NB * 96, 256), 256, 0, stream>>>(pre, scale, shift, hA, NB * 96, 96);

  // ---------------- mlp1 layer 1: hA @ w4 ----------------
  k_zero<<<1, 256, 0, stream>>>(st, 256);
  k_pack_w<<<(96 * 96 + 255) / 256, 256, 0, stream>>>(w4, wpack, 96, 96, 96);
  k_gemm_relu_stats<2, 6, 96, 0, 96><<<gblk, 256, 0, stream>>>(
      hA, nullptr, wpack, b4, pre, colsum, colsumsq, N);
  k_bn_finalize<<<1, 128, 0, stream>>>(colsum, colsumsq, g4, be4, scale, shift, 96, invN);
  k_bn_apply<<<gblocks(NB * 96, 256), 256, 0, stream>>>(pre, scale, shift, hB, NB * 96, 96);

  // ---------------- mlp1 layer 2: hB @ w5 -> d_out ----------------
  k_zero<<<1, 256, 0, stream>>>(st, 256);
  k_pack_w<<<(16 * 96 + 255) / 256, 256, 0, stream>>>(w5, wpack, 96, C, 16);
  k_gemm_relu_stats<2, 1, 96, 0, 10><<<gblk, 256, 0, stream>>>(
      hB, nullptr, wpack, b5, pre, colsum, colsumsq, N);
  k_bn_finalize<<<1, 128, 0, stream>>>(colsum, colsumsq, g5, be5, scale, shift, C, invN);
  k_bn_apply<<<gblocks(NB * C, 256), 256, 0, stream>>>(pre, scale, shift, (float*)d_out,
                                                       NB * C, C);
}
